// SplineConv_83090437308571
// MI455X (gfx1250) — compile-verified, matrix operand order checked by reading, not run on hardware
//
#include <hip/hip_runtime.h>
#include <hip/hip_bf16.h>
#include <stdint.h>

// ---------------------------------------------------------------------------
// Types for CDNA5 WMMA (wave32, v_wmma_f32_16x16x32_bf16)
// ---------------------------------------------------------------------------
typedef __hip_bfloat16 bf16;
typedef __attribute__((ext_vector_type(16))) __bf16 v16bf;
typedef __attribute__((ext_vector_type(8)))  float  v8f;

union FragU { uint4 u[2]; v16bf v; };

#define KSPLINE 5   // 5x5 = 25 kernels, degree-1 open B-spline

// ---------------------------------------------------------------------------
// Degree-1 open B-spline basis (matches reference _basis)
// ---------------------------------------------------------------------------
__device__ __forceinline__ void spline_basis(float px, float py,
                                             float basis[4], int wi[4]) {
  float v0 = px * (KSPLINE - 1), v1 = py * (KSPLINE - 1);
  float f0 = floorf(v0), f1 = floorf(v1);
  float r0 = v0 - f0, r1 = v1 - f1;
  int i0 = (int)f0, i1 = (int)f1;
  float c0[2] = {1.0f - r0, r0};
  float c1[2] = {1.0f - r1, r1};
  int   j0[2] = {i0 % KSPLINE, (i0 + 1) % KSPLINE};
  int   j1[2] = {i1 % KSPLINE, (i1 + 1) % KSPLINE};
#pragma unroll
  for (int s1 = 0; s1 < 2; ++s1)
#pragma unroll
    for (int s0 = 0; s0 < 2; ++s0) {
      basis[s0 + 2 * s1] = c0[s0] * c1[s1];
      wi[s0 + 2 * s1]    = j0[s0] + KSPLINE * j1[s1];
    }
}

// ---------------------------------------------------------------------------
// 1) in-degree (float atomics), then in-place reciprocal clamp
// ---------------------------------------------------------------------------
__global__ void deg_kernel(const int* __restrict__ ei, float* __restrict__ deg, int E) {
  int e = blockIdx.x * blockDim.x + threadIdx.x;
  if (e >= E) return;
  atomicAdd(&deg[ei[e]], 1.0f);
}

__global__ void inv_kernel(float* __restrict__ deg, int N) {
  int n = blockIdx.x * blockDim.x + threadIdx.x;
  if (n >= N) return;
  deg[n] = 1.0f / fmaxf(deg[n], 1.0f);
}

// ---------------------------------------------------------------------------
// 2) Layer-1 edge scatter: A[row, wi, i] += basis * x[col, i]   (A: [N,75] f32)
// ---------------------------------------------------------------------------
__global__ void edge1_kernel(const float* __restrict__ pseudo,
                             const int* __restrict__ ei,
                             const float* __restrict__ x,
                             float* __restrict__ A, int E) {
  int e = blockIdx.x * blockDim.x + threadIdx.x;
  if (e >= E) return;
  float2 p = ((const float2*)pseudo)[e];
  float basis[4]; int wi[4];
  spline_basis(p.x, p.y, basis, wi);
  int r = ei[e], c = ei[E + e];
  float x0 = x[3 * c + 0], x1 = x[3 * c + 1], x2 = x[3 * c + 2];
  float* Ar = A + (size_t)r * 75;
#pragma unroll
  for (int s = 0; s < 4; ++s) {
    float b = basis[s];
    int o = wi[s] * 3;
    atomicAdd(Ar + o + 0, b * x0);
    atomicAdd(Ar + o + 1, b * x1);
    atomicAdd(Ar + o + 2, b * x2);
  }
}

// ---------------------------------------------------------------------------
// 3) Ahat[N,96] bf16 = [ A * inv_deg | x | 0-pad ]   (K padded 78 -> 96)
// ---------------------------------------------------------------------------
__global__ void convA_kernel(const float* __restrict__ A,
                             const float* __restrict__ invdeg,
                             const float* __restrict__ x,
                             bf16* __restrict__ Ahat, int N) {
  int idx = blockIdx.x * blockDim.x + threadIdx.x;
  if (idx >= N * 96) return;
  int n = idx / 96, c = idx % 96;
  float v = 0.0f;
  if (c < 75)       v = A[(size_t)n * 75 + c] * invdeg[n];
  else if (c < 78)  v = x[3 * n + (c - 75)];
  Ahat[idx] = __float2bfloat16(v);
}

// ---------------------------------------------------------------------------
// B-fragment packing: buffer index = ((kt*NT + nt)*32 + lane)*16 + j
// lane holds column nt*16 + lane%16; j<8 -> K = kt*32 + half*8 + j,
// j>=8 -> K = kt*32 + 16 + half*8 + (j-8)   (ISA 7.12.2 16-bit layout)
// ---------------------------------------------------------------------------
__device__ __forceinline__ int frag_k(int kt, int lane, int j) {
  int half = lane >> 4;
  return kt * 32 + ((j < 8) ? (half * 8 + j) : (16 + half * 8 + (j - 8)));
}

// B1ext [96 x 1024]: rows 0..74 = weight[25,3,1024] flat, 75..77 = root_weight
__global__ void pack_b1(const float* __restrict__ W, const float* __restrict__ R,
                        bf16* __restrict__ B1f) {
  int idx = blockIdx.x * blockDim.x + threadIdx.x;
  if (idx >= 3 * 64 * 32 * 16) return;
  int j = idx & 15, lane = (idx >> 4) & 31, nt = (idx >> 9) & 63, kt = idx >> 15;
  int ncol = nt * 16 + (lane & 15);
  int kk = frag_k(kt, lane, j);
  float v = 0.0f;
  if (kk < 75)      v = W[(size_t)kk * 1024 + ncol];
  else if (kk < 78) v = R[(size_t)(kk - 75) * 1024 + ncol];
  B1f[idx] = __float2bfloat16(v);
}

// B2ext [1024 x 80]: cols 0..74 = weight1[25,1024,3] (col = k*3+o),
// cols 75..77 = root_weight1, 78..79 = 0
__global__ void pack_b2(const float* __restrict__ W1, const float* __restrict__ R1,
                        bf16* __restrict__ B2f) {
  int idx = blockIdx.x * blockDim.x + threadIdx.x;
  if (idx >= 32 * 5 * 32 * 16) return;
  int j = idx & 15, lane = (idx >> 4) & 31;
  int rest = idx >> 9;
  int nt = rest % 5, kt = rest / 5;
  int ncol = nt * 16 + (lane & 15);
  int kk = frag_k(kt, lane, j);
  float v = 0.0f;
  if (ncol < 75) {
    int k = ncol / 3, o = ncol % 3;
    v = W1[((size_t)k * 1024 + kk) * 3 + o];
  } else if (ncol < 78) {
    v = R1[(size_t)kk * 3 + (ncol - 75)];
  }
  B2f[idx] = __float2bfloat16(v);
}

// ---------------------------------------------------------------------------
// 4) GEMM1: H[N,1024] bf16 = Ahat[N,96] @ B1ext[96,1024]
//    Each wave: one 16-row M-tile x four 16-col N-tiles, 3 k-steps.
// ---------------------------------------------------------------------------
__global__ __launch_bounds__(256) void gemm1_kernel(const bf16* __restrict__ Ahat,
                                                    const bf16* __restrict__ B1f,
                                                    bf16* __restrict__ H, int MT) {
  const int wave = blockIdx.x * (blockDim.x >> 5) + (threadIdx.x >> 5);
  if (wave >= MT * 16) return;
  const int lane = threadIdx.x & 31;
  const int mt = wave >> 4;       // M tile
  const int ng = wave & 15;       // group of 4 N tiles
  const int half = lane >> 4;
  const int m0 = mt * 16;
  const int mrow = m0 + (lane & 15);

  v8f acc[4];
#pragma unroll
  for (int t = 0; t < 4; ++t) acc[t] = (v8f){0.f,0.f,0.f,0.f,0.f,0.f,0.f,0.f};

#pragma unroll
  for (int kt = 0; kt < 3; ++kt) {
    const bf16* ap = Ahat + (size_t)mrow * 96 + kt * 32 + half * 8;
    FragU fa;
    fa.u[0] = *(const uint4*)(ap);        // K chunk [half*8 .. +7]
    fa.u[1] = *(const uint4*)(ap + 16);   // K chunk [16+half*8 .. +7]
#pragma unroll
    for (int t = 0; t < 4; ++t) {
      int nt = ng * 4 + t;
      const bf16* bp = B1f + (((size_t)kt * 64 + nt) * 32 + lane) * 16;
      FragU fb;
      fb.u[0] = *(const uint4*)(bp);
      fb.u[1] = *(const uint4*)(bp + 8);
      acc[t] = __builtin_amdgcn_wmma_f32_16x16x32_bf16(
          false, fa.v, false, fb.v, (short)0, acc[t], false, false);
    }
  }

#pragma unroll
  for (int t = 0; t < 4; ++t) {
    int n0 = (ng * 4 + t) * 16 + (lane & 15);
#pragma unroll
    for (int r = 0; r < 8; ++r) {
      int mr = m0 + r + 8 * half;          // C/D layout: vgpr r -> row r (+8 hi half)
      H[(size_t)mr * 1024 + n0] = __float2bfloat16(acc[t][r]);
    }
  }
}

// ---------------------------------------------------------------------------
// 5) GEMM2: C2[N,80] f32 = H[N,1024] @ B2ext[1024,80]
//    Each wave: one M-tile x all 5 N-tiles (H streamed exactly once), 32 k-steps.
// ---------------------------------------------------------------------------
__global__ __launch_bounds__(256) void gemm2_kernel(const bf16* __restrict__ H,
                                                    const bf16* __restrict__ B2f,
                                                    float* __restrict__ C2, int MT) {
  const int wave = blockIdx.x * (blockDim.x >> 5) + (threadIdx.x >> 5);
  if (wave >= MT) return;
  const int lane = threadIdx.x & 31;
  const int half = lane >> 4;
  const int m0 = wave * 16;
  const int mrow = m0 + (lane & 15);

  v8f acc[5];
#pragma unroll
  for (int t = 0; t < 5; ++t) acc[t] = (v8f){0.f,0.f,0.f,0.f,0.f,0.f,0.f,0.f};

  for (int kt = 0; kt < 32; ++kt) {
    const bf16* ap = H + (size_t)mrow * 1024 + kt * 32 + half * 8;
    FragU fa;
    fa.u[0] = *(const uint4*)(ap);
    fa.u[1] = *(const uint4*)(ap + 16);
#pragma unroll
    for (int t = 0; t < 5; ++t) {
      const bf16* bp = B2f + (((size_t)kt * 5 + t) * 32 + lane) * 16;
      FragU fb;
      fb.u[0] = *(const uint4*)(bp);
      fb.u[1] = *(const uint4*)(bp + 8);
      acc[t] = __builtin_amdgcn_wmma_f32_16x16x32_bf16(
          false, fa.v, false, fb.v, (short)0, acc[t], false, false);
    }
  }

#pragma unroll
  for (int t = 0; t < 5; ++t) {
    int n0 = t * 16 + (lane & 15);
#pragma unroll
    for (int r = 0; r < 8; ++r) {
      int mr = m0 + r + 8 * half;
      C2[(size_t)mr * 80 + n0] = acc[t][r];
    }
  }
}

// ---------------------------------------------------------------------------
// 6) Layer-2 edge gather/scatter: msg = sum_s basis*C2[col, wi*3+o] -> agg[row]
// ---------------------------------------------------------------------------
__global__ void edge2_kernel(const float* __restrict__ pseudo1,
                             const int* __restrict__ ei,
                             const float* __restrict__ C2,
                             float* __restrict__ agg, int E) {
  int e = blockIdx.x * blockDim.x + threadIdx.x;
  if (e >= E) return;
  float2 p = ((const float2*)pseudo1)[e];
  float basis[4]; int wi[4];
  spline_basis(p.x, p.y, basis, wi);
  int r = ei[e], c = ei[E + e];
  const float* Cc = C2 + (size_t)c * 80;
  float m0 = 0.f, m1 = 0.f, m2 = 0.f;
#pragma unroll
  for (int s = 0; s < 4; ++s) {
    float b = basis[s];
    int o = wi[s] * 3;
    m0 += b * Cc[o + 0];
    m1 += b * Cc[o + 1];
    m2 += b * Cc[o + 2];
  }
  atomicAdd(&agg[3 * r + 0], m0);
  atomicAdd(&agg[3 * r + 1], m1);
  atomicAdd(&agg[3 * r + 2], m2);
}

// ---------------------------------------------------------------------------
// 7) Final: out = agg * inv_deg + (h @ root1)   (root term sits in C2 cols 75..77)
// ---------------------------------------------------------------------------
__global__ void final_kernel(const float* __restrict__ agg,
                             const float* __restrict__ C2,
                             const float* __restrict__ invdeg,
                             float* __restrict__ out, int N) {
  int n = blockIdx.x * blockDim.x + threadIdx.x;
  if (n >= N) return;
  float id = invdeg[n];
  const float* Cn = C2 + (size_t)n * 80 + 75;
#pragma unroll
  for (int o = 0; o < 3; ++o)
    out[3 * n + o] = agg[3 * n + o] * id + Cn[o];
}

// ---------------------------------------------------------------------------
// Launch
// ---------------------------------------------------------------------------
extern "C" void kernel_launch(void* const* d_in, const int* in_sizes, int n_in,
                              void* d_out, int out_size, void* d_ws, size_t ws_size,
                              hipStream_t stream) {
  const float* x       = (const float*)d_in[0];
  const float* pseudo  = (const float*)d_in[1];
  const float* pseudo1 = (const float*)d_in[2];
  const float* weight  = (const float*)d_in[3];
  const float* rootw   = (const float*)d_in[4];
  const float* weight1 = (const float*)d_in[5];
  const float* rootw1  = (const float*)d_in[6];
  const int*   ei      = (const int*)d_in[7];

  const int N = in_sizes[0] / 3;      // 50000 (multiple of 16)
  const int E = in_sizes[1] / 2;      // 800000
  const int MT = N / 16;              // 3125 M-tiles

  // Workspace carve (256B aligned slabs)
  char* ws = (char*)d_ws;
  size_t off = 0;
  auto carve = [&](size_t bytes) -> void* {
    void* p = ws + off;
    off = (off + bytes + 255) & ~(size_t)255;
    return p;
  };
  float* invdeg = (float*)carve((size_t)N * 4);              // deg -> inv_deg
  float* A      = (float*)carve((size_t)N * 75 * 4);
  bf16*  Ahat   = (bf16*) carve((size_t)N * 96 * 2);
  bf16*  B1f    = (bf16*) carve((size_t)3 * 64 * 32 * 16 * 2);
  bf16*  H      = (bf16*) carve((size_t)N * 1024 * 2);
  bf16*  B2f    = (bf16*) carve((size_t)32 * 5 * 32 * 16 * 2);
  float* C2     = (float*)carve((size_t)N * 80 * 4);
  float* agg    = (float*)carve((size_t)N * 3 * 4);
  (void)ws_size;

  (void)hipMemsetAsync(invdeg, 0, (size_t)N * 4, stream);
  (void)hipMemsetAsync(A,      0, (size_t)N * 75 * 4, stream);
  (void)hipMemsetAsync(agg,    0, (size_t)N * 3 * 4, stream);

  const int B = 256;
  pack_b1<<<(3 * 64 * 32 * 16 + B - 1) / B, B, 0, stream>>>(weight, rootw, B1f);
  pack_b2<<<(32 * 5 * 32 * 16 + B - 1) / B, B, 0, stream>>>(weight1, rootw1, B2f);

  deg_kernel<<<(E + B - 1) / B, B, 0, stream>>>(ei, invdeg, E);
  inv_kernel<<<(N + B - 1) / B, B, 0, stream>>>(invdeg, N);
  edge1_kernel<<<(E + B - 1) / B, B, 0, stream>>>(pseudo, ei, x, A, E);
  convA_kernel<<<(N * 96 + B - 1) / B, B, 0, stream>>>(A, invdeg, x, Ahat, N);

  gemm1_kernel<<<MT * 2, B, 0, stream>>>(Ahat, B1f, H, MT);        // MT*16 waves
  gemm2_kernel<<<(MT + 7) / 8, B, 0, stream>>>(H, B2f, C2, MT);    // MT waves

  edge2_kernel<<<(E + B - 1) / B, B, 0, stream>>>(pseudo1, ei, C2, agg, E);
  final_kernel<<<(N + B - 1) / B, B, 0, stream>>>(agg, C2, invdeg, (float*)d_out, N);
}
